// GLP_VQVAE_18872086298972
// MI455X (gfx1250) — matrix operand order repaired and checked by reading, not compiled, not run on hardware
//
#include <hip/hip_runtime.h>
#include <hip/hip_fp16.h>
#include <stdint.h>

typedef _Float16 h16;
typedef h16  v8h  __attribute__((ext_vector_type(8)));
typedef h16  v16h __attribute__((ext_vector_type(16)));
typedef float v8f __attribute__((ext_vector_type(8)));
typedef uint32_t u32x4 __attribute__((ext_vector_type(4)));
typedef uint32_t u32x8 __attribute__((ext_vector_type(8)));

union H16x2 { v16h v; struct { v8h lo, hi; } p; };

__device__ __forceinline__ v8f wmma32(const v16h& a, const v16h& b, const v8f& c) {
    // D = A(16x32 f16) * B(32x16 f16) + C(16x16 f32)
    return __builtin_amdgcn_wmma_f32_16x16x32_f16(false, a, false, b, (short)0, c, false, false);
}

// CDNA5 async copy: global -> LDS, 16 bytes, tracked on ASYNCcnt.
__device__ __forceinline__ void async_cp16(uint32_t lds_byte_off, const void* gaddr) {
    asm volatile("global_load_async_to_lds_b128 %0, %1, off"
                 :: "v"(lds_byte_off), "v"(gaddr) : "memory");
}
__device__ __forceinline__ void wait_async0() {
    asm volatile("s_wait_asynccnt 0x0" ::: "memory");
}

// CDNA5 Tensor Data Mover: DMA a contiguous run of 2-byte elements into LDS.
// D# built per ISA 8.3/8.4: count=1, type=2 ("image"), data_size=2B,
// 2D tile with tile_dim1=1; VADDR2/VADDR3 omitted (<=2D tensor).
__device__ __forceinline__ void tdm_load_1d_to_lds(const void* gsrc,
                                                   uint32_t lds_byte_off,
                                                   uint32_t nelem /* u16 elems */)
{
    uint64_t ga = (uint64_t)gsrc;
    u32x4 g0;
    g0[0] = 1u;                                                // count=1 (valid user D#)
    g0[1] = lds_byte_off;                                      // lds_addr
    g0[2] = (uint32_t)ga;                                      // global_addr[31:0]
    g0[3] = (uint32_t)((ga >> 32) & 0x01FFFFFFu) | (2u << 30); // addr[56:32] | type=2
    u32x8 g1;
    g1[0] = 1u << 16;                                          // data_size=1 (2 bytes)
    g1[1] = (nelem & 0xFFFFu) << 16;                           // tensor_dim0[15:0]
    g1[2] = (nelem >> 16) | (1u << 16);                        // tensor_dim0[31:16] | tensor_dim1=1
    g1[3] = (nelem & 0xFFFFu) << 16;                           // tile_dim0
    g1[4] = 1u;                                                // tile_dim1=1, tile_dim2=0
    g1[5] = nelem;                                             // tensor_dim0_stride[31:0]
    g1[6] = 0u;
    g1[7] = 0u;
    asm volatile("tensor_load_to_lds %0, %1" :: "s"(g0), "s"(g1) : "memory");
}

__device__ __forceinline__ v8h zero8h() { v8h v; for (int i = 0; i < 8; ++i) v[i] = (h16)0.f; return v; }

// ---------------------------------------------------------------------------
// Generic implicit-GEMM conv: NHWC(C=32*CINCH, f16) -> NHWC32 f16.
// Block = 256 threads (8 wave32), tile = 8 out-rows x 16 out-cols.
// Wave w owns one output row of 16 pixels x 32 out-channels (2 WMMA N-tiles).
// Input tile staged in LDS via async-to-LDS copies (zeros for halo OOB).
// ---------------------------------------------------------------------------
template<int K, int S, int P, int CINCH>
__global__ __launch_bounds__(256) void conv_wmma_k(
    const h16* __restrict__ in, const h16* __restrict__ wp,
    h16* __restrict__ out, int inH, int inW, int outW)
{
    constexpr int TH = 8, TW = 16;
    constexpr int IH = (TH - 1) * S + K;
    constexpr int IW = (TW - 1) * S + K;
    constexpr int CSTR = CINCH * 32;                 // halves per pixel
    __shared__ __align__(16) h16 tile[IH * IW * CSTR];

    const int n  = blockIdx.z;
    const int y0 = blockIdx.y * TH;
    const int x0 = blockIdx.x * TW;
    const int iy0 = y0 * S - P, ix0 = x0 * S - P;
    const h16* inb = in + (size_t)n * inH * inW * CSTR;

    // ---- stage input tile (16B granules; async where in-range, zero halo)
    const int tid = threadIdx.x;
    constexpr int PC = 4 * CINCH;                    // 16B chunks per pixel
    constexpr int CHUNKS = IH * IW * PC;
    for (int t = tid; t < CHUNKS; t += 256) {
        int pix = t / PC;
        int rem = t - pix * PC;
        int iy = pix / IW, ix = pix - iy * IW;
        int gy = iy0 + iy, gx = ix0 + ix;
        int hoff = pix * CSTR + rem * 8;             // half index in LDS
        if (gy >= 0 && gy < inH && gx >= 0 && gx < inW) {
            async_cp16((uint32_t)(hoff * 2),
                       inb + ((size_t)gy * inW + gx) * CSTR + rem * 8);
        } else {
            *(v8h*)&tile[hoff] = zero8h();
        }
    }
    wait_async0();
    __syncthreads();

    // ---- WMMA accumulation over taps (and Cin chunks)
    const int wave = tid >> 5;
    const int lane = tid & 31;
    const int m = lane & 15, g = lane >> 4;
    v8f acc0 = {}, acc1 = {};
    const int rbase = wave * S;

    for (int ky = 0; ky < K; ++ky) {
#pragma unroll
        for (int kx = 0; kx < K; ++kx) {
#pragma unroll
            for (int c = 0; c < CINCH; ++c) {
                const h16* tp = &tile[((rbase + ky) * IW + (m * S + kx)) * CSTR + c * 32];
                H16x2 a;
                a.p.lo = *(const v8h*)(tp + g * 8);
                a.p.hi = *(const v8h*)(tp + 16 + g * 8);
                const h16* wb = wp + ((size_t)(((ky * K + kx) * CINCH + c) * 2) * 32 + lane) * 16;
                H16x2 b0; b0.p.lo = *(const v8h*)wb;          b0.p.hi = *(const v8h*)(wb + 8);
                const h16* wb1 = wb + 32 * 16;
                H16x2 b1; b1.p.lo = *(const v8h*)wb1;         b1.p.hi = *(const v8h*)(wb1 + 8);
                acc0 = wmma32(a.v, b0.v, acc0);
                acc1 = wmma32(a.v, b1.v, acc1);
            }
        }
    }

    // ---- store D (lane holds N=m, M=r+8*g rows of the 16-pixel tile)
    const int outH = (int)gridDim.y * TH;
    const int oy = y0 + wave;
    h16* ob = out + ((size_t)(n * outH + oy) * outW + x0) * 32;
#pragma unroll
    for (int r = 0; r < 8; ++r) {
        int M = r + 8 * g;
        ob[(size_t)M * 32 + m]      = (h16)acc0[r];
        ob[(size_t)M * 32 + 16 + m] = (h16)acc1[r];
    }
}

// ---------------------------------------------------------------------------
// ConvTranspose2d k=4 s=2 p=1 via WMMA direct gather.
// Block tile: 4 out-rows x 32 out-cols; wave -> (row, x-parity) so the 2x2
// valid taps are wave-uniform (B fragment uniform; EXEC all-1 for WMMA).
// ---------------------------------------------------------------------------
__global__ __launch_bounds__(256) void convt_wmma_k(
    const h16* __restrict__ in, const h16* __restrict__ wp,
    h16* __restrict__ out, int inH, int inW)
{
    const int outW = inW * 2;
    const int n  = blockIdx.z;
    const int y0 = blockIdx.y * 4, x0 = blockIdx.x * 32;
    const int tid = threadIdx.x, wave = tid >> 5, lane = tid & 31;
    const int m = lane & 15, g = lane >> 4;
    const int row = wave >> 1, par = wave & 1;
    const int oy = y0 + row;
    const int kyb = (oy + 1) & 1;
    const int kxb = 1 - par;
    const h16* inb = in + (size_t)n * inH * inW * 32;
    v8f acc0 = {}, acc1 = {};

#pragma unroll
    for (int a_ = 0; a_ < 2; ++a_) {
        const int ky = kyb + 2 * a_;
        const int iy = ((oy + 1) >> 1) - a_;
#pragma unroll
        for (int b_ = 0; b_ < 2; ++b_) {
            const int kx = kxb + 2 * b_;
            const int ix = (x0 >> 1) + m + par - b_;     // per-lane pixel column
            H16x2 a;
            if (iy >= 0 && iy < inH && ix >= 0 && ix < inW) {
                const h16* tp = inb + ((size_t)iy * inW + ix) * 32;
                a.p.lo = *(const v8h*)(tp + g * 8);
                a.p.hi = *(const v8h*)(tp + 16 + g * 8);
            } else {
                a.p.lo = zero8h(); a.p.hi = zero8h();
            }
            const int tap = ky * 4 + kx;
            const h16* wb = wp + ((size_t)(tap * 2) * 32 + lane) * 16;
            H16x2 b0; b0.p.lo = *(const v8h*)wb;  b0.p.hi = *(const v8h*)(wb + 8);
            const h16* wb1 = wb + 32 * 16;
            H16x2 b1; b1.p.lo = *(const v8h*)wb1; b1.p.hi = *(const v8h*)(wb1 + 8);
            acc0 = wmma32(a.v, b0.v, acc0);
            acc1 = wmma32(a.v, b1.v, acc1);
        }
    }

    h16* ob = out + ((size_t)(n * (inH * 2) + oy) * outW) * 32;
#pragma unroll
    for (int r = 0; r < 8; ++r) {
        int M = r + 8 * g;
        int ox = x0 + 2 * M + par;
        ob[(size_t)ox * 32 + m]      = (h16)acc0[r];
        ob[(size_t)ox * 32 + 16 + m] = (h16)acc1[r];
    }
}

// ---------------------------------------------------------------------------
// Pack conv weights (f32 OIHW, or IOHW for transposed) into B-fragment order:
// dst[(((tap*CINCH + c)*2 + nt)*32 + lane)*16 + j], lane holds (N=lane%16,
// K chunk = (j/8)*16 + (lane/16)*8 + j%8).  Cin slot s maps to real input ch (s/32)*24+s%32.
// ---------------------------------------------------------------------------
__global__ void pack_conv_w_k(const float* __restrict__ w, h16* __restrict__ dst,
                              int O, int I, int KK, int CINCH, int transposed)
{
    int total = KK * CINCH * 2 * 32 * 16;
    for (int idx = blockIdx.x * blockDim.x + threadIdx.x; idx < total;
         idx += gridDim.x * blockDim.x) {
        int j    = idx & 15;
        int lane = (idx >> 4) & 31;
        int nt   = (idx >> 9) & 1;
        int rest = idx >> 10;
        int c    = rest % CINCH;
        int tap  = rest / CINCH;
        int o  = nt * 16 + (lane & 15);
        int kk = ((j >> 3) << 4) + ((lane >> 4) << 3) + (j & 7);
        int s  = c * 32 + kk;
        int rr = s & 31;
        int i  = (s >> 5) * 24 + rr;
        float v = 0.f;
        if (rr < 24 && i < I && o < O)
            v = transposed ? w[((size_t)i * O + o) * KK + tap]
                           : w[((size_t)o * I + i) * KK + tap];
        dst[idx] = (h16)v;
    }
}

// embed [24,256] f32 -> code [256][32] f16 (+8 zero pad) and |e|^2 norms
__global__ void pack_embed_k(const float* __restrict__ e, h16* __restrict__ code,
                             float* __restrict__ norms)
{
    int nidx = blockIdx.x * blockDim.x + threadIdx.x;
    if (nidx >= 256) return;
    float s = 0.f;
    for (int d = 0; d < 32; ++d) {
        float v = (d < 24) ? e[(size_t)d * 256 + nidx] : 0.f;
        code[(size_t)nidx * 32 + d] = (h16)v;
        s += v * v;
    }
    norms[nidx] = s;
}

// VQ: argmin_n (|e_n|^2 - 2 f.e_n), gather code, accumulate sum (q-f)^2.
// Codebook (16 KiB) is DMA'd into LDS by the Tensor Data Mover (one wave
// issues tensor_load_to_lds; TENSORcnt waited before the block barrier).
__global__ __launch_bounds__(256) void quantize_k(
    const h16* __restrict__ in, const h16* __restrict__ code,
    const float* __restrict__ norms, h16* __restrict__ qt,
    float* __restrict__ acc, int npix)
{
    __shared__ __align__(16) h16 cb[256 * 32];   // must stay first: LDS offset 0
    __shared__ float nb[256];
    __shared__ float red[256];

    if (threadIdx.x < 32) {                      // wave 0 drives the TDM
        tdm_load_1d_to_lds(code, 0u, 256 * 32);
        __builtin_amdgcn_s_wait_tensorcnt(0);
    }
    nb[threadIdx.x] = norms[threadIdx.x];
    __syncthreads();

    float local = 0.f;
    for (int p = blockIdx.x * blockDim.x + threadIdx.x; p < npix;
         p += gridDim.x * blockDim.x) {
        float f[24];
        const h16* fp = in + (size_t)p * 32;
#pragma unroll
        for (int d = 0; d < 24; ++d) f[d] = (float)fp[d];
        int best = 0; float bestd = 3.4e38f;
        for (int c = 0; c < 256; ++c) {
            const h16* cp = &cb[c * 32];
            float dot = 0.f;
#pragma unroll
            for (int d = 0; d < 24; ++d) dot += f[d] * (float)cp[d];
            float dd = nb[c] - 2.f * dot;
            if (dd < bestd) { bestd = dd; best = c; }
        }
        const h16* cp = &cb[best * 32];
        h16* qp = qt + (size_t)p * 32;
#pragma unroll
        for (int d = 0; d < 32; ++d) qp[d] = cp[d];
#pragma unroll
        for (int d = 0; d < 24; ++d) { float df = (float)cp[d] - f[d]; local += df * df; }
    }
    red[threadIdx.x] = local;
    __syncthreads();
    for (int s = 128; s > 0; s >>= 1) {
        if ((int)threadIdx.x < s) red[threadIdx.x] += red[threadIdx.x + s];
        __syncthreads();
    }
    if (threadIdx.x == 0) atomicAdd(acc, red[0]);
}

// channel mean / max maps (2 x f32 per pixel)
__global__ void meanmax_k(const h16* __restrict__ in, float* __restrict__ mm, int npix)
{
    int p = blockIdx.x * blockDim.x + threadIdx.x;
    if (p >= npix) return;
    const h16* fp = in + (size_t)p * 32;
    float s = 0.f, mx = -3.4e38f;
#pragma unroll
    for (int d = 0; d < 24; ++d) { float v = (float)fp[d]; s += v; mx = fmaxf(mx, v); }
    mm[(size_t)p * 2 + 0] = s * (1.f / 24.f);
    mm[(size_t)p * 2 + 1] = mx;
}

// out = base + relu(h2 * sigmoid(conv7x7(meanmax, w_sa)))   (fused residual)
__global__ void attn_res_k(const h16* __restrict__ h2, const float* __restrict__ mm,
                           const float* __restrict__ wsa, const h16* __restrict__ base,
                           h16* __restrict__ out, int H, int W, int npix)
{
    int p = blockIdx.x * blockDim.x + threadIdx.x;
    if (p >= npix) return;
    int HW = H * W;
    int n = p / HW, r = p - n * HW, y = r / W, x = r - y * W;
    const float* mmb = mm + (size_t)n * HW * 2;
    float s = 0.f;
    for (int ky = 0; ky < 7; ++ky) {
        int yy = y + ky - 3; if (yy < 0 || yy >= H) continue;
        for (int kx = 0; kx < 7; ++kx) {
            int xx = x + kx - 3; if (xx < 0 || xx >= W) continue;
            const float* q = &mmb[((size_t)yy * W + xx) * 2];
            s += q[0] * wsa[ky * 7 + kx] + q[1] * wsa[49 + ky * 7 + kx];
        }
    }
    float sig = 1.f / (1.f + __expf(-s));
    const h16* hp = h2 + (size_t)p * 32;
    const h16* bp = base + (size_t)p * 32;
    h16* op = out + (size_t)p * 32;
#pragma unroll
    for (int d = 0; d < 32; ++d) {
        float v = (float)hp[d] * sig;
        v = v > 0.f ? v : 0.f;
        op[d] = (h16)((float)bp[d] + v);
    }
}

// concat two NHWC32 tensors into NHWC64
__global__ void concat_k(const h16* __restrict__ a, const h16* __restrict__ b,
                         h16* __restrict__ out, int npix)
{
    int t = blockIdx.x * blockDim.x + threadIdx.x;
    int total = npix * 8;
    if (t >= total) return;
    int p = t >> 3, q = t & 7;
    const h16* src = (q < 4) ? (a + (size_t)p * 32 + q * 8)
                             : (b + (size_t)p * 32 + (q - 4) * 8);
    *(v8h*)(out + (size_t)p * 64 + q * 8) = *(const v8h*)src;
}

// grayscale mean over 3 channels (NCHW input)
__global__ void gray_k(const float* __restrict__ x, float* __restrict__ g,
                       int HW, int npix)
{
    int p = blockIdx.x * blockDim.x + threadIdx.x;
    if (p >= npix) return;
    int n = p / HW, r = p - n * HW;
    const float* xb = x + (size_t)n * 3 * HW + r;
    g[p] = (xb[0] + xb[HW] + xb[2 * (size_t)HW]) * (1.f / 3.f);
}

// Laplacian 3x3 (pad 1) + 1x1 conv 1->24 into NHWC32 f16; keep xl f32
__global__ void lap_in_k(const float* __restrict__ g, const float* __restrict__ win,
                         float* __restrict__ xl, h16* __restrict__ h0,
                         int H, int W, int npix)
{
    int p = blockIdx.x * blockDim.x + threadIdx.x;
    if (p >= npix) return;
    int HW = H * W;
    int n = p / HW, r = p - n * HW, y = r / W, x = r - y * W;
    const float* gb = g + (size_t)n * HW;
    float c  = gb[r];
    float up = (y > 0)     ? gb[r - W] : 0.f;
    float dn = (y < H - 1) ? gb[r + W] : 0.f;
    float lf = (x > 0)     ? gb[r - 1] : 0.f;
    float rt = (x < W - 1) ? gb[r + 1] : 0.f;
    float l = 4.f * c - up - dn - lf - rt;
    xl[p] = l;
    h16* hp = h0 + (size_t)p * 32;
#pragma unroll
    for (int d = 0; d < 32; ++d) hp[d] = (h16)(d < 24 ? l * win[d] : 0.f);
}

// mask = 1x1 conv 24->1 ; clean = xl - mask
__global__ void final_k(const h16* __restrict__ dec1, const float* __restrict__ wout,
                        const float* __restrict__ xl, float* __restrict__ omask,
                        float* __restrict__ oclean, int npix)
{
    int p = blockIdx.x * blockDim.x + threadIdx.x;
    if (p >= npix) return;
    const h16* dp = dec1 + (size_t)p * 32;
    float s = 0.f;
#pragma unroll
    for (int d = 0; d < 24; ++d) s += (float)dp[d] * wout[d];
    omask[p] = s;
    oclean[p] = xl[p] - s;
}

__global__ void zero_k(float* p, int n)
{
    int i = blockIdx.x * blockDim.x + threadIdx.x;
    if (i < n) p[i] = 0.f;
}

__global__ void diff_final_k(const float* __restrict__ acc, float* __restrict__ out,
                             float inv2, float inv3)
{
    if (blockIdx.x == 0 && threadIdx.x == 0)
        out[0] = acc[0] * inv2 + acc[1] * inv3;
}

// ---------------------------------------------------------------------------
extern "C" void kernel_launch(void* const* d_in, const int* in_sizes, int n_in,
                              void* d_out, int out_size, void* d_ws, size_t ws_size,
                              hipStream_t stream)
{
    (void)in_sizes; (void)n_in; (void)out_size; (void)ws_size;

    // setup_inputs order: x, then sorted names
    const float* x         = (const float*)d_in[0];
    const float* embed2    = (const float*)d_in[1];
    const float* embed3    = (const float*)d_in[2];
    const float* w_img_in  = (const float*)d_in[3];
    const float* w_img_out = (const float*)d_in[4];
    const float* w_out_[6]; // dd,dm,dt,ed,em,et
    const float* w_pw_[6];
    const float* w_res_[6];
    const float* w_sa_[6];
    for (int i = 0; i < 6; ++i) {
        w_out_[i] = (const float*)d_in[5 + i];
        w_pw_[i]  = (const float*)d_in[11 + i];
        w_res_[i] = (const float*)d_in[17 + i];
        w_sa_[i]  = (const float*)d_in[23 + i];
    }
    const float* w_squ1 = (const float*)d_in[29];
    const float* w_squ2 = (const float*)d_in[30];

    const int N = 8;
    const int HWF = 512 * 512, HWH = 256 * 256, HWQ = 128 * 128, HWE = 64 * 64;

    size_t off = 0;
    auto alloc = [&](size_t bytes) -> void* {
        void* p = (char*)d_ws + off;
        off += (bytes + 255) & ~(size_t)255;
        return p;
    };
    float* G   = (float*)alloc((size_t)N * HWF * 4);
    float* XL  = (float*)alloc((size_t)N * HWF * 4);
    h16* F0    = (h16*)alloc((size_t)N * HWF * 32 * 2);
    h16* F1    = (h16*)alloc((size_t)N * HWF * 32 * 2);
    h16* F2    = (h16*)alloc((size_t)N * HWF * 32 * 2);
    float* MMF = (float*)alloc((size_t)N * HWF * 2 * 4);
    h16* ENC1  = (h16*)alloc((size_t)N * HWH * 32 * 2);
    h16* Hb1   = (h16*)alloc((size_t)N * HWH * 32 * 2);
    h16* Hb2   = (h16*)alloc((size_t)N * HWH * 32 * 2);
    h16* Hb3   = (h16*)alloc((size_t)N * HWH * 32 * 2);
    h16* CCH   = (h16*)alloc((size_t)N * HWH * 64 * 2);
    float* MMH = (float*)alloc((size_t)N * HWH * 2 * 4);
    h16* ENC2  = (h16*)alloc((size_t)N * HWQ * 32 * 2);
    h16* Qb1   = (h16*)alloc((size_t)N * HWQ * 32 * 2);
    h16* Qb2   = (h16*)alloc((size_t)N * HWQ * 32 * 2);
    h16* Qb3   = (h16*)alloc((size_t)N * HWQ * 32 * 2);
    h16* QT2   = (h16*)alloc((size_t)N * HWQ * 32 * 2);
    h16* CCQ   = (h16*)alloc((size_t)N * HWQ * 64 * 2);
    float* MMQ = (float*)alloc((size_t)N * HWQ * 2 * 4);
    h16* ENC3  = (h16*)alloc((size_t)N * HWE * 32 * 2);
    h16* QT3   = (h16*)alloc((size_t)N * HWE * 32 * 2);

    auto wbytes = [](int KK, int CINCH) { return (size_t)KK * CINCH * 2 * 32 * 16 * 2; };
    const int kres[6] = {3, 5, 7, 3, 5, 7};  // dd,dm,dt,ed,em,et
    h16* P_res[6]; h16* P_pw[6]; h16* P_out[6];
    for (int i = 0; i < 6; ++i) {
        P_res[i] = (h16*)alloc(wbytes(kres[i] * kres[i], 1));
        P_pw[i]  = (h16*)alloc(wbytes(1, 1));
        P_out[i] = (h16*)alloc(wbytes(16, 1));
    }
    h16* P_squ1 = (h16*)alloc(wbytes(1, 2));
    h16* P_squ2 = (h16*)alloc(wbytes(1, 2));
    h16* CODE2 = (h16*)alloc(256 * 32 * 2);
    h16* CODE3 = (h16*)alloc(256 * 32 * 2);
    float* NRM2 = (float*)alloc(256 * 4);
    float* NRM3 = (float*)alloc(256 * 4);
    float* DACC = (float*)alloc(2 * 4);

    auto ew = [](int n) { return dim3((unsigned)((n + 255) / 256)); };
    auto pack = [&](const float* w, h16* dst, int O, int I, int KK, int CINCH, int tr) {
        int total = KK * CINCH * 2 * 32 * 16;
        pack_conv_w_k<<<dim3((unsigned)((total + 255) / 256)), 256, 0, stream>>>(
            w, dst, O, I, KK, CINCH, tr);
    };

    // ---- pack weights / codebooks, zero diff accumulators
    for (int i = 0; i < 6; ++i) {
        pack(w_res_[i], P_res[i], 24, 24, kres[i] * kres[i], 1, 0);
        pack(w_pw_[i],  P_pw[i],  24, 24, 1, 1, 0);
        pack(w_out_[i], P_out[i], 24, 24, 16, 1, (i < 3) ? 1 : 0);  // dec uses IOHW
    }
    pack(w_squ1, P_squ1, 24, 48, 1, 2, 0);
    pack(w_squ2, P_squ2, 24, 48, 1, 2, 0);
    pack_embed_k<<<1, 256, 0, stream>>>(embed2, CODE2, NRM2);
    pack_embed_k<<<1, 256, 0, stream>>>(embed3, CODE3, NRM3);
    zero_k<<<1, 32, 0, stream>>>(DACC, 2);

    // ---- input stage: gray mean -> Laplacian -> 1x1 expand to 24ch
    gray_k<<<ew(N * HWF), 256, 0, stream>>>(x, G, HWF, N * HWF);
    lap_in_k<<<ew(N * HWF), 256, 0, stream>>>(G, w_img_in, XL, F0, 512, 512, N * HWF);

    // ---- encoder: et (full, k7)
    conv_wmma_k<7,1,3,1><<<dim3(32, 64, N), 256, 0, stream>>>(F0, P_res[5], F1, 512, 512, 512);
    conv_wmma_k<1,1,0,1><<<dim3(32, 64, N), 256, 0, stream>>>(F1, P_pw[5],  F2, 512, 512, 512);
    meanmax_k<<<ew(N * HWF), 256, 0, stream>>>(F2, MMF, N * HWF);
    attn_res_k<<<ew(N * HWF), 256, 0, stream>>>(F2, MMF, w_sa_[5], F0, F1, 512, 512, N * HWF);
    conv_wmma_k<4,2,1,1><<<dim3(16, 32, N), 256, 0, stream>>>(F1, P_out[5], ENC1, 512, 512, 256);

    // ---- encoder: em (half, k5)
    conv_wmma_k<5,1,2,1><<<dim3(16, 32, N), 256, 0, stream>>>(ENC1, P_res[4], Hb1, 256, 256, 256);
    conv_wmma_k<1,1,0,1><<<dim3(16, 32, N), 256, 0, stream>>>(Hb1, P_pw[4], Hb2, 256, 256, 256);
    meanmax_k<<<ew(N * HWH), 256, 0, stream>>>(Hb2, MMH, N * HWH);
    attn_res_k<<<ew(N * HWH), 256, 0, stream>>>(Hb2, MMH, w_sa_[4], ENC1, Hb1, 256, 256, N * HWH);
    conv_wmma_k<4,2,1,1><<<dim3(8, 16, N), 256, 0, stream>>>(Hb1, P_out[4], ENC2, 256, 256, 128);

    // ---- encoder: ed (quarter, k3)
    conv_wmma_k<3,1,1,1><<<dim3(8, 16, N), 256, 0, stream>>>(ENC2, P_res[3], Qb1, 128, 128, 128);
    conv_wmma_k<1,1,0,1><<<dim3(8, 16, N), 256, 0, stream>>>(Qb1, P_pw[3], Qb2, 128, 128, 128);
    meanmax_k<<<ew(N * HWQ), 256, 0, stream>>>(Qb2, MMQ, N * HWQ);
    attn_res_k<<<ew(N * HWQ), 256, 0, stream>>>(Qb2, MMQ, w_sa_[3], ENC2, Qb1, 128, 128, N * HWQ);
    conv_wmma_k<4,2,1,1><<<dim3(4, 8, N), 256, 0, stream>>>(Qb1, P_out[3], ENC3, 128, 128, 64);

    // ---- vector quantization (+ codebook-diff partial sums)
    quantize_k<<<ew(N * HWQ), 256, 0, stream>>>(ENC2, CODE2, NRM2, QT2, DACC + 0, N * HWQ);
    quantize_k<<<ew(N * HWE), 256, 0, stream>>>(ENC3, CODE3, NRM3, QT3, DACC + 1, N * HWE);

    // ---- decoder: dd (eighth -> quarter, k3)
    convt_wmma_k<<<dim3(4, 32, N), 256, 0, stream>>>(QT3, P_out[0], Qb3, 64, 64);
    conv_wmma_k<3,1,1,1><<<dim3(8, 16, N), 256, 0, stream>>>(Qb3, P_res[0], Qb1, 128, 128, 128);
    conv_wmma_k<1,1,0,1><<<dim3(8, 16, N), 256, 0, stream>>>(Qb1, P_pw[0], Qb2, 128, 128, 128);
    meanmax_k<<<ew(N * HWQ), 256, 0, stream>>>(Qb2, MMQ, N * HWQ);
    attn_res_k<<<ew(N * HWQ), 256, 0, stream>>>(Qb2, MMQ, w_sa_[0], Qb3, Qb1, 128, 128, N * HWQ);
    concat_k<<<ew(N * HWQ * 8), 256, 0, stream>>>(Qb1, QT2, CCQ, N * HWQ);
    conv_wmma_k<1,1,0,2><<<dim3(8, 16, N), 256, 0, stream>>>(CCQ, P_squ2, Qb3, 128, 128, 128);

    // ---- decoder: dm (quarter -> half, k5)
    convt_wmma_k<<<dim3(8, 64, N), 256, 0, stream>>>(Qb3, P_out[1], Hb3, 128, 128);
    conv_wmma_k<5,1,2,1><<<dim3(16, 32, N), 256, 0, stream>>>(Hb3, P_res[1], Hb1, 256, 256, 256);
    conv_wmma_k<1,1,0,1><<<dim3(16, 32, N), 256, 0, stream>>>(Hb1, P_pw[1], Hb2, 256, 256, 256);
    meanmax_k<<<ew(N * HWH), 256, 0, stream>>>(Hb2, MMH, N * HWH);
    attn_res_k<<<ew(N * HWH), 256, 0, stream>>>(Hb2, MMH, w_sa_[1], Hb3, Hb1, 256, 256, N * HWH);
    concat_k<<<ew(N * HWH * 8), 256, 0, stream>>>(Hb1, ENC1, CCH, N * HWH);
    conv_wmma_k<1,1,0,2><<<dim3(16, 32, N), 256, 0, stream>>>(CCH, P_squ1, Hb3, 256, 256, 256);

    // ---- decoder: dt (half -> full, k7)
    convt_wmma_k<<<dim3(16, 128, N), 256, 0, stream>>>(Hb3, P_out[2], F0, 256, 256);
    conv_wmma_k<7,1,3,1><<<dim3(32, 64, N), 256, 0, stream>>>(F0, P_res[2], F1, 512, 512, 512);
    conv_wmma_k<1,1,0,1><<<dim3(32, 64, N), 256, 0, stream>>>(F1, P_pw[2], F2, 512, 512, 512);
    meanmax_k<<<ew(N * HWF), 256, 0, stream>>>(F2, MMF, N * HWF);
    attn_res_k<<<ew(N * HWF), 256, 0, stream>>>(F2, MMF, w_sa_[2], F0, F1, 512, 512, N * HWF);

    // ---- head: mask = 1x1(dec1), clean = xl - mask, diff scalar
    float* omask  = (float*)d_out;
    float* oclean = omask + (size_t)N * HWF;
    final_k<<<ew(N * HWF), 256, 0, stream>>>(F1, w_img_out, XL, omask, oclean, N * HWF);
    diff_final_k<<<1, 32, 0, stream>>>(DACC, omask + (size_t)2 * N * HWF,
                                       1.f / (float)(N * HWQ * 24),
                                       1.f / (float)(N * HWE * 24));
}
